// MyModel_7876970021378
// MI455X (gfx1250) — matrix operand order repaired and checked by reading, not compiled, not run on hardware
//
#include <hip/hip_runtime.h>
#include <hip/hip_bf16.h>

typedef __attribute__((ext_vector_type(16))) _Float16 v16h;
typedef __attribute__((ext_vector_type(8)))  _Float16 v8h;
typedef __attribute__((ext_vector_type(8)))  float    v8f;

#define N_EXPERTS        3
#define IN_COLS          35
#define FRAGS_PER_EXPERT 17
#define TOTAL_FRAGS      (N_EXPERTS * FRAGS_PER_EXPERT)   // 51
#define FRAG_HALFS       512                              // 32 lanes x 16 halfs
#define WAVES            4

struct PtrTable { const float* p[97]; };

// ---------------------------------------------------------------------------
// Pack weights into CDNA5 WMMA B-fragment layout (16x16x32 f16):
//   B is 32x16 (KxN). lane n (<16) holds col n, halfs0-7 = K0..7, halfs8-15 = K16..23;
//   lane n+16 holds col n, halfs0-7 = K8..15, halfs8-15 = K24..31.  Zero-pad K to 32, N to 16.
// Biases: padded 16-float vectors per fragment (column n -> bias[n]).
// ---------------------------------------------------------------------------
__global__ void pack_params(PtrTable pt, _Float16* __restrict__ wfrag,
                            float* __restrict__ wbias)
{
    // local frag -> W leaf index within expert (32 leaves: W,b per layer,
    // towers in order t5,t4,t3,t2,t1,integ), dims, N-tile base
    constexpr int LEAFW[FRAGS_PER_EXPERT] = {0,0,2,4,6,8, 10,12,14,16, 18,20,22, 24,26, 28, 30};
    constexpr int DIN [FRAGS_PER_EXPERT]  = {32,32,32,16,8,4, 32,16,8,4, 32,8,4, 32,4, 32, 5};
    constexpr int DOUT[FRAGS_PER_EXPERT]  = {32,32,16,8,4,1, 16,8,4,1, 8,4,1, 4,1, 1, 1};
    constexpr int NB  [FRAGS_PER_EXPERT]  = {0,16,0,0,0,0, 0,0,0,0, 0,0,0, 0,0, 0, 0};

    const int tid    = blockIdx.x * blockDim.x + threadIdx.x;
    const int stride = gridDim.x * blockDim.x;

    for (int idx = tid; idx < TOTAL_FRAGS * FRAG_HALFS; idx += stride) {
        int fg   = idx >> 9;        // fragment
        int w    = idx & 511;
        int lane = w >> 4;
        int h    = w & 15;
        int e    = fg / FRAGS_PER_EXPERT;
        int lf   = fg % FRAGS_PER_EXPERT;
        int k    = ((h < 8) ? h : h + 8) + ((lane < 16) ? 0 : 8);
        int n    = NB[lf] + (lane & 15);
        const float* W = pt.p[1 + e * 32 + LEAFW[lf]];
        float v = (k < DIN[lf] && n < DOUT[lf]) ? W[k * DOUT[lf] + n] : 0.0f;
        wfrag[idx] = (_Float16)v;
    }
    for (int idx = tid; idx < TOTAL_FRAGS * 16; idx += stride) {
        int fg = idx >> 4;
        int n0 = idx & 15;
        int e  = fg / FRAGS_PER_EXPERT;
        int lf = fg % FRAGS_PER_EXPERT;
        int n  = NB[lf] + n0;
        const float* B = pt.p[1 + e * 32 + LEAFW[lf] + 1];
        wbias[idx] = (n < DOUT[lf]) ? B[n] : 0.0f;
    }
}

// ---------------------------------------------------------------------------
// Branch-free mish:  tanh(softplus(x)) = (t^2+2t)/(t^2+2t+2), t = e^x.
//   mish(x) = x * u * rcp(u+2),  u = t*(t+2)
// Clamp x at 40 so u stays finite; ratio saturates to 1 -> mish(x) = x.
// x -> -inf: t -> 0, result 0.  One v_exp_f32 + one v_rcp_f32, no branches.
// ---------------------------------------------------------------------------
__device__ __forceinline__ float mish_f(float x)
{
    float xc = fminf(x, 40.0f);
    float t  = __builtin_amdgcn_exp2f(xc * 1.4426950408889634f);  // e^x
    float u  = t * (t + 2.0f);
    return x * u * __builtin_amdgcn_rcpf(u + 2.0f);
}

// A fragment (16x32 f16, MxK): lane m(<16): halfs0-7 = K0..7, halfs8-15 = K16..23
//                              lane m+16  : halfs0-7 = K8..15, halfs8-15 = K24..31
__device__ __forceinline__ v16h load_A(const _Float16* tile, int lane)
{
    const int m  = lane & 15;
    const int c0 = (lane < 16) ? 0 : 8;
    const _Float16* p = tile + m * 32;
    v8h lo = *(const v8h*)(p + c0);        // ds_load_b128
    v8h hi = *(const v8h*)(p + c0 + 16);   // ds_load_b128
    v16h a;
#pragma unroll
    for (int i = 0; i < 8; ++i) { a[i] = lo[i]; a[8 + i] = hi[i]; }
    return a;
}

__device__ __forceinline__ v16h load_Bfrag(const _Float16* wfrag, int fg, int lane)
{
    return *(const v16h*)(wfrag + fg * FRAG_HALFS + lane * 16);  // 32B, cached
}

__device__ __forceinline__ void zero_tile(_Float16* tile, int lane)
{
    v8h z;
#pragma unroll
    for (int i = 0; i < 8; ++i) z[i] = (_Float16)0.0f;
    *(v8h*)(tile + lane * 16)     = z;
    *(v8h*)(tile + lane * 16 + 8) = z;
}

__global__ __launch_bounds__(WAVES * 32)
void moe_forward(const float* __restrict__ in, const _Float16* __restrict__ wfrag,
                 const float* __restrict__ wbias, float* __restrict__ out, int N)
{
    __shared__ __align__(16) _Float16 s_tiles[WAVES][4][512];   // X,Y0,Y1,H per wave
    __shared__ float s_cls[WAVES][48];
    __shared__ float s_res[WAVES][N_EXPERTS * 16];

    const int  lane = threadIdx.x & 31;
    const int  wv   = threadIdx.x >> 5;
    const long tile = (long)blockIdx.x * WAVES + wv;
    const long base = tile * 16;
    if (base >= N) return;

    _Float16* tptr[4] = { s_tiles[wv][0], s_tiles[wv][1],
                          s_tiles[wv][2], s_tiles[wv][3] };
    _Float16* tX = tptr[0];
    _Float16* tH = tptr[3];

    // Coalesced load of the 16x35 input block (f16 features + f32 class cols).
    for (int i = lane; i < 16 * IN_COLS; i += 32) {
        int  r   = i / IN_COLS;
        int  c   = i - r * IN_COLS;
        long row = base + r;
        float v  = (row < N) ? in[row * IN_COLS + c] : 0.0f;
        if (c < 32) tX[r * 32 + c] = (_Float16)v;
        else        s_cls[wv][r * 3 + (c - 32)] = v;
    }

    // argmax over the 3 class columns (first-max semantics, like jnp.argmax)
    const int myrow = lane & 15;
    float a0 = s_cls[wv][myrow * 3 + 0];
    float a1 = s_cls[wv][myrow * 3 + 1];
    float a2 = s_cls[wv][myrow * 3 + 2];
    int cls = 0; float bv = a0;
    if (a1 > bv) { cls = 1; bv = a1; }
    if (a2 > bv) { cls = 2; }

    // Per-expert layer schedule (16 layers): src tile, dst tile, #N-tiles,
    // fragment offset, H-column (-1 = normal, 0..4 = tower output, -2 = result)
    constexpr int SRC [16] = {0,1,2,1,2, 0,1,2,1, 0,1,2, 0,1, 0, 3};
    constexpr int DSTT[16] = {1,2,1,2,0, 1,2,1,0, 1,2,0, 1,0, 0, 0};
    constexpr int NT  [16] = {2,1,1,1,1, 1,1,1,1, 1,1,1, 1,1, 1, 1};
    constexpr int FRG [16] = {0,2,3,4,5, 6,7,8,9, 10,11,12, 13,14, 15, 16};
    constexpr int HC  [16] = {-1,-1,-1,-1,0, -1,-1,-1,1, -1,-1,2, -1,3, 4, -2};

    const int nlane = lane & 15;
    const int m0    = (lane < 16) ? 0 : 8;

    for (int e = 0; e < N_EXPERTS; ++e) {
        const int fbase = e * FRAGS_PER_EXPERT;
        zero_tile(tH, lane);
#pragma unroll
        for (int L = 0; L < 16; ++L) {
            v16h a = load_A(tptr[SRC[L]], lane);
            if (HC[L] == -1) {
                _Float16* dst = tptr[DSTT[L]];
                zero_tile(dst, lane);            // keep K-padding exactly zero
#pragma unroll
                for (int t = 0; t < NT[L]; ++t) {
                    const int fg = fbase + FRG[L] + t;
                    v16h  b  = load_Bfrag(wfrag, fg, lane);
                    float bn = wbias[fg * 16 + nlane];
                    v8f c;
#pragma unroll
                    for (int r = 0; r < 8; ++r) c[r] = bn;   // bias as accum init
                    c = __builtin_amdgcn_wmma_f32_16x16x32_f16(
                            false, a, false, b, (short)0, c, false, false);
#pragma unroll
                    for (int r = 0; r < 8; ++r)
                        dst[(m0 + r) * 32 + t * 16 + nlane] = (_Float16)mish_f(c[r]);
                }
            } else {
                const int fg = fbase + FRG[L];
                v16h  b  = load_Bfrag(wfrag, fg, lane);
                float bn = wbias[fg * 16 + nlane];
                v8f c;
#pragma unroll
                for (int r = 0; r < 8; ++r) c[r] = bn;
                c = __builtin_amdgcn_wmma_f32_16x16x32_f16(
                        false, a, false, b, (short)0, c, false, false);
                if (nlane == 0) {                // column 0 = lanes 0 and 16
                    if (HC[L] >= 0) {
#pragma unroll
                        for (int r = 0; r < 8; ++r)
                            tH[(m0 + r) * 32 + HC[L]] = (_Float16)mish_f(c[r]);
                    } else {                     // integ layer -> expert result
#pragma unroll
                        for (int r = 0; r < 8; ++r)
                            s_res[wv][e * 16 + m0 + r] = mish_f(c[r]);
                    }
                }
            }
        }
    }

    long row = base + myrow;
    if (lane < 16 && row < N)
        out[row] = s_res[wv][cls * 16 + myrow];
}

// ---------------------------------------------------------------------------
extern "C" void kernel_launch(void* const* d_in, const int* in_sizes, int n_in,
                              void* d_out, int out_size, void* d_ws, size_t ws_size,
                              hipStream_t stream)
{
    PtrTable pt;
    for (int i = 0; i < 97; ++i)
        pt.p[i] = (i < n_in) ? (const float*)d_in[i] : nullptr;

    _Float16* wfrag = (_Float16*)d_ws;
    float*    wbias = (float*)((char*)d_ws + (size_t)TOTAL_FRAGS * FRAG_HALFS * sizeof(_Float16));

    const float* in = (const float*)d_in[0];
    const int N = in_sizes[0] / IN_COLS;

    pack_params<<<52, 256, 0, stream>>>(pt, wfrag, wbias);

    const int tiles  = (N + 15) / 16;
    const int blocks = (tiles + WAVES - 1) / WAVES;
    moe_forward<<<blocks, WAVES * 32, 0, stream>>>(in, wfrag, wbias, (float*)d_out, N);
}
